// ChunkwiseCausalAttention_43250320671025
// MI455X (gfx1250) — compile-verified, matrix operand order checked by reading, not run on hardware
//
#include <hip/hip_runtime.h>

#define HIDDEN 1024
#define HEADS 16
#define HDIM 64
#define BATCH 2
#define SEQ 4096
#define TOKENS (BATCH * SEQ) // 8192

typedef __attribute__((ext_vector_type(16))) __bf16 v16bf;
typedef __attribute__((ext_vector_type(8)))  float  v8f;

union FragAB { unsigned int u[8]; v16bf v; };
union FragC  { float f[8];        v8f   v; };

__device__ __forceinline__ unsigned short f2bf(float x) {
    unsigned int u = __float_as_uint(x);
    u += 0x7FFFu + ((u >> 16) & 1u);   // round-to-nearest-even
    return (unsigned short)(u >> 16);
}
__device__ __forceinline__ unsigned int packbf(float lo, float hi) {
    return (unsigned int)f2bf(lo) | ((unsigned int)f2bf(hi) << 16);
}
// K-striping of 16-bit A/B WMMA fragments (CDNA5 ISA 7.12.2):
// vgpr v holds k-pair starting at kbase(v, lane>>4).
__device__ __forceinline__ int kbase(int v, int half) {
    return 2 * v + ((v >= 4) ? 8 : 0) + half * 8;
}

// ---------- conversion kernels ----------
__global__ void __launch_bounds__(256) cvt_x_kernel(const float* __restrict__ x,
                                                    unsigned int* __restrict__ xb) {
    int gid = blockIdx.x * 256 + threadIdx.x;              // one bf16 pair per thread
    if (gid < TOKENS * (HIDDEN / 2))
        xb[gid] = packbf(x[2 * gid], x[2 * gid + 1]);
}

// P[kp][n] = {bf16(W[2kp][n]) , bf16(W[2kp+1][n])}   (k-pair packed for B fragments)
__global__ void __launch_bounds__(256) cvt_pack_kernel(const float* __restrict__ W,
                                                       unsigned int* __restrict__ P,
                                                       int K2, int N) {
    int gid = blockIdx.x * 256 + threadIdx.x;
    if (gid >= K2 * N) return;
    int kp = gid / N, n = gid - kp * N;
    P[gid] = packbf(W[(2 * kp) * N + n], W[(2 * kp + 1) * N + n]);
}

// ---------- QKV projection: [8192,1024] x [1024,3072] ----------
// block = 256 thr (8 waves), wave tile = 16 rows x 64 cols, block tile 128x64.
__global__ void __launch_bounds__(256) qkv_gemm_kernel(
    const unsigned int* __restrict__ xb,   // bf16 pairs [8192][512]
    const unsigned int* __restrict__ wp,   // packed    [512][3072]
    unsigned short* __restrict__ q16,      // [B,H,S,64] bf16
    unsigned short* __restrict__ k16,      // [B,H,S,64] bf16
    unsigned int*  __restrict__ vp)        // [B,H,S/2,64] key-pair packed
{
    const int bid = blockIdx.x;
    const int ct = bid % 48, rt = bid / 48;
    const int tid = threadIdx.x;
    const int w = tid >> 5, lane = tid & 31;
    const int m = lane & 15, half = lane >> 4;
    const int rowbase = rt * 128 + w * 16;
    const int colbase = ct * 64;

    FragC acc[4];
    #pragma unroll
    for (int t = 0; t < 4; t++)
        #pragma unroll
        for (int v = 0; v < 8; v++) acc[t].f[v] = 0.f;

    for (int k0 = 0; k0 < HIDDEN; k0 += 32) {
        FragAB a;
        #pragma unroll
        for (int v = 0; v < 8; v++) {
            int kk = k0 + kbase(v, half);
            a.u[v] = xb[(size_t)(rowbase + m) * 512 + (kk >> 1)];
        }
        #pragma unroll
        for (int t = 0; t < 4; t++) {
            FragAB b;
            #pragma unroll
            for (int v = 0; v < 8; v++) {
                int kk = k0 + kbase(v, half);
                b.u[v] = wp[(size_t)(kk >> 1) * 3072 + colbase + t * 16 + m];
            }
            acc[t].v = __builtin_amdgcn_wmma_f32_16x16x32_bf16(
                false, a.v, false, b.v, (short)0, acc[t].v, false, false);
        }
    }

    #pragma unroll
    for (int t = 0; t < 4; t++) {
        int col = colbase + t * 16 + m;        // lane's output column
        int sel = col >> 10;                   // 0=Q 1=K 2=V
        int rem = col & 1023;
        int hh = rem >> 6, d = rem & 63;
        if (sel == 2) {
            // V: pack consecutive key rows (rows per lane are consecutive: m=v+8*half)
            #pragma unroll
            for (int j = 0; j < 4; j++) {
                int r0 = rowbase + half * 8 + 2 * j;
                int bb = r0 >> 12, ss = r0 & 4095;
                vp[((size_t)(bb * 16 + hh) * 2048 + (ss >> 1)) * 64 + d] =
                    packbf(acc[t].f[2 * j], acc[t].f[2 * j + 1]);
            }
        } else {
            unsigned short* dst = (sel == 0) ? q16 : k16;
            #pragma unroll
            for (int v = 0; v < 8; v++) {
                int r = rowbase + half * 8 + v;
                int bb = r >> 12, ss = r & 4095;
                dst[((size_t)(bb * 16 + hh) * 4096 + ss) * 64 + d] = f2bf(acc[t].f[v]);
            }
        }
    }
}

// ---------- attention: 4 waves/block, wave = 16 query rows, online softmax ----------
__global__ void __launch_bounds__(128) attn_kernel(
    const unsigned int* __restrict__ q32,  // bf16 pairs [B*H*4096][32]
    const unsigned int* __restrict__ k32,  // bf16 pairs [B*H*4096][32]
    const unsigned int* __restrict__ vp,   // packed     [B*H*2048][64]
    unsigned short* __restrict__ ao)       // bf16       [8192][1024]
{
    __shared__ float srow[4 * 16 * 128];   // per-wave private 16x128 score tile
    __shared__ float rstat[4 * 16];        // per-wave row stats scratch

    const int bid = blockIdx.x;
    const int rblk = bid & 63, h = (bid >> 6) & 15, b = bid >> 10;
    const int tid = threadIdx.x;
    const int w = tid >> 5, lane = tid & 31;
    const int m = lane & 15, half = lane >> 4;
    const int qrow0 = rblk * 64 + w * 16;

    const unsigned int* qb = q32 + (size_t)(b * 16 + h) * 4096 * 32;
    const unsigned int* kp = k32 + (size_t)(b * 16 + h) * 4096 * 32;
    const unsigned int* vb = vp  + (size_t)(b * 16 + h) * 2048 * 64;
    float* ps = srow + w * (16 * 128);
    float* pr = rstat + w * 16;

    // Q fragments (persist for whole kernel): d 0..31 and 32..63
    FragAB aq[2];
    #pragma unroll
    for (int s = 0; s < 2; s++)
        #pragma unroll
        for (int v = 0; v < 8; v++)
            aq[s].u[v] = qb[(size_t)(qrow0 + m) * 32 + ((s * 32 + kbase(v, half)) >> 1)];

    FragC of[4];
    #pragma unroll
    for (int t = 0; t < 4; t++)
        #pragma unroll
        for (int v = 0; v < 8; v++) of[t].f[v] = 0.f;
    float mrun[8], lrun[8];
    #pragma unroll
    for (int v = 0; v < 8; v++) { mrun[v] = -1.0e30f; lrun[v] = 0.f; }

    const int nch = ((qrow0 + 15) >> 7) + 1;   // causal: key chunks of 128
    for (int kc = 0; kc < nch; kc++) {
        const int kb0 = kc * 128;
        float cm[8];
        #pragma unroll
        for (int v = 0; v < 8; v++) cm[v] = -1.0e30f;

        // ---- S = Q K^T / 8 with causal mask, 8 key sub-tiles of 16 ----
        for (int kt = 0; kt < 8; kt++) {
            const int key0 = kb0 + kt * 16;
            FragC s;
            #pragma unroll
            for (int v = 0; v < 8; v++) s.f[v] = 0.f;
            #pragma unroll
            for (int ss = 0; ss < 2; ss++) {
                FragAB bk;
                #pragma unroll
                for (int v = 0; v < 8; v++)
                    bk.u[v] = kp[(size_t)(key0 + m) * 32 + ((ss * 32 + kbase(v, half)) >> 1)];
                s.v = __builtin_amdgcn_wmma_f32_16x16x32_bf16(
                    false, aq[ss].v, false, bk.v, (short)0, s.v, false, false);
            }
            #pragma unroll
            for (int v = 0; v < 8; v++) {
                int key = key0 + m;                 // D-frag: lane holds column n=m
                int qg  = qrow0 + half * 8 + v;     // D-frag: row = v + 8*half
                float val = (key <= qg) ? s.f[v] * 0.125f : -1.0e30f;
                cm[v] = fmaxf(cm[v], val);
                ps[(half * 8 + v) * 128 + kt * 16 + m] = val;
            }
        }
        // row max across the 16 lanes of each half (xor<16 stays in-half)
        #pragma unroll
        for (int off = 1; off < 16; off <<= 1)
            #pragma unroll
            for (int v = 0; v < 8; v++)
                cm[v] = fmaxf(cm[v], __shfl_xor(cm[v], off, 32));
        // online-softmax rescale (O-frag rows align with v+8*half exactly)
        #pragma unroll
        for (int v = 0; v < 8; v++) {
            float nm = fmaxf(mrun[v], cm[v]);
            float fc = __expf(mrun[v] - nm);
            mrun[v] = nm;
            lrun[v] *= fc;
            #pragma unroll
            for (int t = 0; t < 4; t++) of[t].f[v] *= fc;
        }
        if (m == 0) {
            #pragma unroll
            for (int v = 0; v < 8; v++) pr[half * 8 + v] = mrun[v];
        }
        __builtin_amdgcn_wave_barrier();
        float rowm = pr[m];

        // ---- P = exp(S - rowmax) rebuilt as A fragments; O += P V ----
        float csum = 0.f;
        for (int c = 0; c < 4; c++) {
            FragAB pa;
            #pragma unroll
            for (int v = 0; v < 8; v++) {
                int kk = c * 32 + kbase(v, half);
                float p0 = __expf(ps[m * 128 + kk]     - rowm);
                float p1 = __expf(ps[m * 128 + kk + 1] - rowm);
                csum += p0 + p1;
                pa.u[v] = packbf(p0, p1);
            }
            #pragma unroll
            for (int t = 0; t < 4; t++) {
                FragAB bv;
                #pragma unroll
                for (int v = 0; v < 8; v++)
                    bv.u[v] = vb[(size_t)((kb0 + c * 32 + kbase(v, half)) >> 1) * 64 + t * 16 + m];
                of[t].v = __builtin_amdgcn_wmma_f32_16x16x32_bf16(
                    false, pa.v, false, bv.v, (short)0, of[t].v, false, false);
            }
        }
        csum += __shfl_xor(csum, 16, 32);   // combine the two halves of each row
        __builtin_amdgcn_wave_barrier();
        if (lane < 16) pr[lane] = csum;
        __builtin_amdgcn_wave_barrier();
        #pragma unroll
        for (int v = 0; v < 8; v++) lrun[v] += pr[half * 8 + v];
    }

    // ---- normalize and store bf16 [token][h*64+d] ----
    #pragma unroll
    for (int t = 0; t < 4; t++)
        #pragma unroll
        for (int v = 0; v < 8; v++) {
            float o = of[t].f[v] / lrun[v];
            int r = qrow0 + half * 8 + v;
            ao[(size_t)(b * 4096 + r) * 1024 + h * 64 + t * 16 + m] = f2bf(o);
        }
}

// ---------- output projection: [8192,1024] x [1024,1024] -> fp32 ----------
__global__ void __launch_bounds__(256) out_gemm_kernel(
    const unsigned int* __restrict__ ab,   // bf16 pairs [8192][512]
    const unsigned int* __restrict__ wp,   // packed     [512][1024]
    float* __restrict__ out)               // fp32       [8192][1024]
{
    const int bid = blockIdx.x;
    const int ct = bid & 15, rt = bid >> 4;
    const int tid = threadIdx.x;
    const int w = tid >> 5, lane = tid & 31;
    const int m = lane & 15, half = lane >> 4;
    const int rowbase = rt * 128 + w * 16;
    const int colbase = ct * 64;

    FragC acc[4];
    #pragma unroll
    for (int t = 0; t < 4; t++)
        #pragma unroll
        for (int v = 0; v < 8; v++) acc[t].f[v] = 0.f;

    for (int k0 = 0; k0 < HIDDEN; k0 += 32) {
        FragAB a;
        #pragma unroll
        for (int v = 0; v < 8; v++) {
            int kk = k0 + kbase(v, half);
            a.u[v] = ab[(size_t)(rowbase + m) * 512 + (kk >> 1)];
        }
        #pragma unroll
        for (int t = 0; t < 4; t++) {
            FragAB b;
            #pragma unroll
            for (int v = 0; v < 8; v++) {
                int kk = k0 + kbase(v, half);
                b.u[v] = wp[(size_t)(kk >> 1) * 1024 + colbase + t * 16 + m];
            }
            acc[t].v = __builtin_amdgcn_wmma_f32_16x16x32_bf16(
                false, a.v, false, b.v, (short)0, acc[t].v, false, false);
        }
    }
    #pragma unroll
    for (int t = 0; t < 4; t++) {
        int col = colbase + t * 16 + m;
        #pragma unroll
        for (int v = 0; v < 8; v++) {
            int r = rowbase + half * 8 + v;
            out[(size_t)r * 1024 + col] = acc[t].f[v];
        }
    }
}

extern "C" void kernel_launch(void* const* d_in, const int* in_sizes, int n_in,
                              void* d_out, int out_size, void* d_ws, size_t ws_size,
                              hipStream_t stream) {
    const float* x    = (const float*)d_in[0];
    const float* wqkv = (const float*)d_in[1];
    const float* wout = (const float*)d_in[2];
    float* out = (float*)d_out;
    char* ws = (char*)d_ws;

    // workspace layout (88 MB total)
    unsigned int*   xb  = (unsigned int*)  (ws + ((size_t) 0 << 20)); // 16 MB bf16 x
    unsigned int*   wqp = (unsigned int*)  (ws + ((size_t)16 << 20)); //  6 MB packed w_qkv
    unsigned int*   wop = (unsigned int*)  (ws + ((size_t)22 << 20)); //  2 MB packed w_out
    unsigned short* q16 = (unsigned short*)(ws + ((size_t)24 << 20)); // 16 MB Q [B,H,S,64]
    unsigned short* k16 = (unsigned short*)(ws + ((size_t)40 << 20)); // 16 MB K [B,H,S,64]
    unsigned int*   vpk = (unsigned int*)  (ws + ((size_t)56 << 20)); // 16 MB V packed
    unsigned short* ao  = (unsigned short*)(ws + ((size_t)72 << 20)); // 16 MB attn out

    cvt_x_kernel   <<<16384, 256, 0, stream>>>(x, xb);
    cvt_pack_kernel<<< 6144, 256, 0, stream>>>(wqkv, wqp, 512, 3072);
    cvt_pack_kernel<<< 2048, 256, 0, stream>>>(wout, wop, 512, 1024);
    qkv_gemm_kernel<<< 3072, 256, 0, stream>>>(xb, wqp, q16, k16, vpk);
    attn_kernel    <<< 2048, 128, 0, stream>>>((const unsigned int*)q16,
                                               (const unsigned int*)k16, vpk, ao);
    out_gemm_kernel<<< 1024, 256, 0, stream>>>((const unsigned int*)ao, wop, out);
}
